// SerializedLHSTransformer_53171695125274
// MI455X (gfx1250) — matrix operand order repaired and checked by reading, not compiled, not run on hardware
//
#include <hip/hip_runtime.h>
#include <math.h>

// ---------------- problem constants (match reference) ----------------
constexpr int BATCH   = 4;
constexpr int NTOK    = 16384;          // points per batch
constexpr int DMODEL  = 256;
constexpr int NHEAD   = 8;
constexpr int DHK     = 32;
constexpr int DHV     = 32;
constexpr int DFF     = 1024;
constexpr int NSAMP   = 128;
constexpr int NPAD    = 16512;          // 129*128
constexpr int LSEQ    = 129;            // local sequence length
constexpr int NSEQ_L  = 512;            // BATCH*128 local sequences
constexpr int MTOK    = BATCH * NPAD;   // 66048 token rows (div by 32)
constexpr int DEPTH   = 32;
constexpr float BIGF  = 1000000000.0f;
constexpr float ATT_SCALE = 0.17677669529663687f;  // 1/sqrt(32)
constexpr int NLAYER  = 5;

typedef _Float16 half_t;
typedef __attribute__((ext_vector_type(16))) _Float16 v16h;
typedef __attribute__((ext_vector_type(8)))  _Float16 v8h;
typedef __attribute__((ext_vector_type(8)))  float    v8f;

#ifndef __has_builtin
#define __has_builtin(x) 0
#endif
#if __has_builtin(__builtin_amdgcn_tensor_load_to_lds) && \
    __has_builtin(__builtin_amdgcn_s_wait_tensorcnt)
#define USE_TDM 1
#else
#define USE_TDM 0
#endif

#if USE_TDM
// ---------------- Tensor Data Mover: 2D f16 tile global -> LDS ----------------
typedef unsigned int tdm_u32x4 __attribute__((ext_vector_type(4)));
typedef int          tdm_i32x4 __attribute__((ext_vector_type(4)));
typedef int          tdm_i32x8 __attribute__((ext_vector_type(8)));

// Loads `rows` rows of `rowElems` f16 elements (row stride `strideElems` elements)
// from global into contiguous LDS at lds_dst. D# layout per CDNA5 ISA.
__device__ inline void tdm_load_tile_f16(const void* lds_dst, const void* global_src,
                                         unsigned rows, unsigned rowElems,
                                         unsigned long long strideElems) {
  unsigned long long ga = (unsigned long long)(uintptr_t)global_src;
  unsigned lds_off = (unsigned)(uintptr_t)lds_dst;   // low 32 bits of generic ptr = LDS offset
  tdm_u32x4 g0;
  g0[0] = 1u;                                        // count=1, user descriptor
  g0[1] = lds_off;                                   // lds_addr (bytes)
  g0[2] = (unsigned)ga;                              // global_addr[31:0]
  g0[3] = (unsigned)((ga >> 32) & 0x01ffffffu) | (2u << 30);  // global_addr[56:32] | type=2
  tdm_i32x8 g1;
  g1[0] = (int)(1u << 16);                           // workgroup_mask=0, data_size=1 (2B)
  g1[1] = (int)((rowElems & 0xffffu) << 16);         // tensor_dim0[15:0]
  g1[2] = (int)(((rowElems >> 16) & 0xffffu) | ((rows & 0xffffu) << 16));  // td0 hi | td1 lo
  g1[3] = (int)(((rows >> 16) & 0xffffu) | ((rowElems & 0xffffu) << 16));  // td1 hi | tile_dim0
  g1[4] = (int)(rows & 0xffffu);                     // tile_dim1 | tile_dim2=0
  g1[5] = (int)(strideElems & 0xffffffffu);          // tensor_dim0_stride lo
  g1[6] = (int)((strideElems >> 32) & 0xffffu);      // tensor_dim0_stride hi | td1_stride=0
  g1[7] = 0;
  tdm_i32x4 gz = {0, 0, 0, 0};
#if __clang_major__ >= 23
  tdm_i32x8 gz8 = {0, 0, 0, 0, 0, 0, 0, 0};
  __builtin_amdgcn_tensor_load_to_lds(g0, g1, gz, gz, gz8, 0);
#else
  __builtin_amdgcn_tensor_load_to_lds(g0, g1, gz, gz, 0);
#endif
}
#endif  // USE_TDM

// ---------------- small device helpers ----------------
__device__ inline unsigned hash4(unsigned a, unsigned b, unsigned c, unsigned d) {
  unsigned h = a * 0x9E3779B1u ^ b * 0x85EBCA77u ^ c * 0xC2B2AE3Du ^ d * 0x27D4EB2Fu;
  h ^= h >> 15; h *= 0x2C1B3C6Du; h ^= h >> 12; h *= 0x297A2D39u; h ^= h >> 15;
  return h;
}
__device__ inline float u01f(unsigned h) { return (h >> 8) * (1.0f / 16777216.0f); }
__device__ inline float gaussf(unsigned h) {
  float u1 = fmaxf(u01f(hash4(h, 0x68bc21ebu, 1u, 2u)), 1e-7f);
  float u2 = u01f(hash4(h, 0x02e5be93u, 3u, 4u));
  return sqrtf(-2.0f * __logf(u1)) * __cosf(6.28318530718f * u2);
}
__device__ inline float gelu_exact(float v) {
  return 0.5f * v * (1.0f + erff(v * 0.70710678118654752f));
}

// in-LDS bitonic sort (ascending by key), nitems = power of two
__device__ void bitonic_sort(float* key, int* val, int nitems, int tid, int nthreads) {
  for (int k = 2; k <= nitems; k <<= 1) {
    for (int j = k >> 1; j > 0; j >>= 1) {
      __syncthreads();
      for (int i = tid; i < nitems; i += nthreads) {
        int ix = i ^ j;
        if (ix > i) {
          bool up = ((i & k) == 0);
          float a = key[i], b = key[ix];
          bool sw = up ? (a > b) : (a < b);
          if (sw) {
            key[i] = b; key[ix] = a;
            int t = val[i]; val[i] = val[ix]; val[ix] = t;
          }
        }
      }
    }
  }
  __syncthreads();
}

// ---------------- 1. serialization scan (per batch, points live in LDS) ----------------
constexpr int SER_T = 1024;
constexpr int SER_I = NTOK / SER_T;  // 16

__global__ void __launch_bounds__(1024) k_serialize(const float* __restrict__ coords,
                                                    const float* __restrict__ mask,
                                                    float* __restrict__ totalOut) {
  extern __shared__ __align__(16) char smem[];
  float* px  = (float*)smem;
  float* py  = px + NTOK;
  float* pz  = py + NTOK;
  float* red = pz + NTOK;              // [1024]
  int*  redi = (int*)(red + SER_T);    // [1024]

  const int bi  = blockIdx.x;
  const int tid = threadIdx.x;

  float mloc[SER_I], tot[SER_I];
  for (int it = 0; it < SER_I; ++it) {
    int i = tid + it * SER_T;
    size_t gi = (size_t)bi * NTOK + i;
    px[i] = coords[gi * 3 + 0];
    py[i] = coords[gi * 3 + 1];
    pz[i] = coords[gi * 3 + 2];
    mloc[it] = mask[gi];
    tot[it] = 0.0f;
  }
  float sm = 0.0f;
  for (int it = 0; it < SER_I; ++it) sm += mloc[it];
  __syncthreads();
  red[tid] = sm; __syncthreads();
  for (int o = SER_T / 2; o > 0; o >>= 1) { if (tid < o) red[tid] += red[tid + o]; __syncthreads(); }
  const float summask = red[0];
  __syncthreads();

  float o0 = 1.0f, o1 = 1.0f, o2 = 1.0f;  // previous ABC
  for (int step = 0; step < DEPTH; ++step) {
    float scale = (step < 18) ? (float)(131072 >> step) : 0.0f;  // 2^17 >> step
    float A0 = gaussf(hash4(bi, step, 11u, 0u));
    float A1 = gaussf(hash4(bi, step, 12u, 0u));
    float A2 = gaussf(hash4(bi, step, 13u, 0u));
    float inA = rsqrtf(fmaxf(A0 * A0 + A1 * A1 + A2 * A2, 1e-24f));
    float inO = rsqrtf(fmaxf(o0 * o0 + o1 * o1 + o2 * o2, 1e-24f));
    float ang = (A0 * o0 + A1 * o1 + A2 * o2) * inA * inO;
    float sg = (ang < 0.0f) ? -1.0f : 1.0f;  // sign(0) -> +1 per reference
    A0 *= sg; A1 *= sg; A2 *= sg;

    float proj[SER_I]; float ps = 0.0f;
    for (int it = 0; it < SER_I; ++it) {
      int i = tid + it * SER_T;
      proj[it] = px[i] * A0 + py[i] * A1 + pz[i] * A2;
      ps += proj[it] * mloc[it];
    }
    __syncthreads();
    red[tid] = ps; __syncthreads();
    for (int o = SER_T / 2; o > 0; o >>= 1) { if (tid < o) red[tid] += red[tid + o]; __syncthreads(); }
    const float mean = red[0] / (summask + 1e-9f);
    __syncthreads();

    float sc[SER_I];
    for (int it = 0; it < SER_I; ++it) {
      float d = proj[it] - mean;
      sc[it] = (d > 0.0f) ? 1.0f : ((d < 0.0f) ? -1.0f : 0.0f);
      tot[it] += sc[it] * scale;
    }

    float dup[2][3];
    for (int side = 0; side < 2; ++side) {
      float want = side ? 1.0f : -1.0f;
      float d0 = 0.0f, d1 = 0.0f, d2 = 0.0f;
      for (int p = 0; p < 2; ++p) {
        float best = 3.0e38f; int bix = tid;
        for (int it = 0; it < SER_I; ++it) {
          int i = tid + it * SER_T;
          float pen = (sc[it] == want && mloc[it] > 0.0f) ? 0.0f : BIGF;
          float r = u01f(hash4(bi, step, (unsigned)(side * 2 + p + 100), (unsigned)i)) + pen;
          if (r < best) { best = r; bix = i; }
        }
        red[tid] = best; redi[tid] = bix; __syncthreads();
        for (int o = SER_T / 2; o > 0; o >>= 1) {
          if (tid < o && red[tid + o] < red[tid]) { red[tid] = red[tid + o]; redi[tid] = redi[tid + o]; }
          __syncthreads();
        }
        int wsel = redi[0];
        d0 += px[wsel]; d1 += py[wsel]; d2 += pz[wsel];
        __syncthreads();
      }
      dup[side][0] = d0 * 0.5f; dup[side][1] = d1 * 0.5f; dup[side][2] = d2 * 0.5f;
    }
    for (int it = 0; it < SER_I; ++it) {
      int i = tid + it * SER_T;
      float ml = (sc[it] < 0.0f && mloc[it] > 0.0f) ? 1.0f : 0.0f;
      float mr = (sc[it] > 0.0f && mloc[it] > 0.0f) ? 1.0f : 0.0f;
      float nx = (px[i] - dup[0][0]) * ml + (px[i] - dup[1][0]) * mr;
      float ny = (py[i] - dup[0][1]) * ml + (py[i] - dup[1][1]) * mr;
      float nz = (pz[i] - dup[0][2]) * ml + (pz[i] - dup[1][2]) * mr;
      px[i] = nx; py[i] = ny; pz[i] = nz;
    }
    __syncthreads();
    o0 = A0; o1 = A1; o2 = A2;
  }
  for (int it = 0; it < SER_I; ++it) {
    int i = tid + it * SER_T;
    totalOut[(size_t)bi * NTOK + i] = tot[it] + 524288.0f * (1.0f - mloc[it]);  // 2^19 = inf
  }
}

// ---------------- 2. full argsort of 16384 keys per batch (LDS bitonic) ----------------
__global__ void __launch_bounds__(1024) k_argsort16k(const float* __restrict__ total,
                                                     int* __restrict__ idxOut) {
  extern __shared__ __align__(16) char smem[];
  float* key = (float*)smem;
  int*   val = (int*)(key + NTOK);
  const int bi = blockIdx.x;
  for (int i = threadIdx.x; i < NTOK; i += 1024) {
    key[i] = total[(size_t)bi * NTOK + i];
    val[i] = i;
  }
  __syncthreads();
  bitonic_sort(key, val, NTOK, threadIdx.x, 1024);
  for (int i = threadIdx.x; i < NTOK; i += 1024) idxOut[(size_t)bi * NTOK + i] = val[i];
}

__global__ void k_rev(const int* __restrict__ idx, int* __restrict__ rev) {
  int t = blockIdx.x * 256 + threadIdx.x;
  if (t >= BATCH * NTOK) return;
  int bi = t / NTOK, i = t % NTOK;
  rev[bi * NTOK + idx[t]] = i;
}

// ---------------- 3. gather + pad into (BATCH, NPAD, 256) ----------------
__global__ void k_gather_pad(const float* __restrict__ x, const float* __restrict__ mask,
                             const int* __restrict__ idx, float* __restrict__ xo,
                             float* __restrict__ mo) {
  int row = blockIdx.x;                    // 0..MTOK-1
  int bi = row / NPAD, r = row % NPAD;
  float4* dst = (float4*)(xo + (size_t)row * DMODEL);
  if (r < NTOK) {
    int t = idx[bi * NTOK + r];
    const float4* src = (const float4*)(x + ((size_t)bi * NTOK + t) * DMODEL);
    for (int j = threadIdx.x; j < DMODEL / 4; j += blockDim.x) dst[j] = src[j];
    if (threadIdx.x == 0) mo[row] = mask[bi * NTOK + t];
  } else {
    for (int j = threadIdx.x; j < DMODEL / 4; j += blockDim.x) dst[j] = make_float4(0, 0, 0, 0);
    if (threadIdx.x == 0) mo[row] = 0.0f;
  }
}

// ---------------- 4. RMSNorm: f32 row -> f16 row (one wave per token) ----------------
__global__ void k_rmsnorm(const float* __restrict__ x, const float* __restrict__ g,
                          half_t* __restrict__ out, int M) {
  int wave = (blockIdx.x * blockDim.x + threadIdx.x) >> 5;
  int lane = threadIdx.x & 31;
  if (wave >= M) return;
  const float* px = x + (size_t)wave * DMODEL;
  float v[8]; float ss = 0.0f;
  for (int j = 0; j < 8; ++j) { v[j] = px[lane + j * 32]; ss += v[j] * v[j]; }
  for (int o = 16; o > 0; o >>= 1) ss += __shfl_xor(ss, o, 32);
  float r = rsqrtf(ss * (1.0f / DMODEL) + 1e-6f);
  half_t* po = out + (size_t)wave * DMODEL;
  for (int j = 0; j < 8; ++j) po[lane + j * 32] = (half_t)(v[j] * r * g[lane + j * 32]);
}

// ---------------- 5. weight transpose f32[K][N] -> f16[N][K] ----------------
__global__ void k_transpose_h(const float* __restrict__ src, half_t* __restrict__ dst,
                              int K, int Nn) {
  int t = blockIdx.x * 256 + threadIdx.x;
  if (t >= K * Nn) return;
  int n = t % Nn, k = t / Nn;
  dst[(size_t)n * K + k] = (half_t)src[(size_t)k * Nn + n];
}

// ---------------- 6. WMMA GEMM: C[M][N] = A[M][K](f16) * Bt[N][K](f16) + bias ----------------
// Tiles staged via Tensor Data Mover when available. Each wave: 16x32 output
// (2 accumulators, A-fragment reused for 2 WMMAs per k-step).
// mode 0: out f16 = C ; mode 1: out f16 = gelu(C) ; mode 2: outF f32 += C * rs
constexpr int GBM = 32, GBN = 128, GKT = 128;

__global__ void __launch_bounds__(256) k_gemm(const half_t* __restrict__ A,
                                              const half_t* __restrict__ Bt,
                                              const float* __restrict__ bias,
                                              int M, int Nn, int K, int mode,
                                              const float* __restrict__ rs,
                                              half_t* __restrict__ outH,
                                              float* __restrict__ outF) {
  __shared__ half_t As[GBM][GKT];   //  8 KB
  __shared__ half_t Bs[GBN][GKT];   // 32 KB
  const int bm = blockIdx.x * GBM, bn = blockIdx.y * GBN;
  const int tid = threadIdx.x;
  const int wave = tid >> 5, lane = tid & 31;
  const int wm = wave >> 2;              // 0..1 : 16-row slice
  const int wnb = (wave & 3) * 32;       // 0,32,64,96 : 32-col slice
  const int g = lane >> 4, ln = lane & 15;
  v8f c0 = {}, c1 = {};
  for (int k0 = 0; k0 < K; k0 += GKT) {
    __syncthreads();   // previous tiles fully consumed
#if USE_TDM
    if (wave == 0) {
      tdm_load_tile_f16(&As[0][0], A + (size_t)bm * K + k0, GBM, GKT, (unsigned long long)K);
      tdm_load_tile_f16(&Bs[0][0], Bt + (size_t)bn * K + k0, GBN, GKT, (unsigned long long)K);
      __builtin_amdgcn_s_wait_tensorcnt(0);
    }
#else
    for (int c8 = tid; c8 < (GBM * GKT / 8); c8 += 256) {
      int r = c8 / (GKT / 8); int cc = (c8 % (GKT / 8)) * 8;
      *(v8h*)&As[r][cc] = *(const v8h*)(A + (size_t)(bm + r) * K + k0 + cc);
    }
    for (int c8 = tid; c8 < (GBN * GKT / 8); c8 += 256) {
      int r = c8 / (GKT / 8); int cc = (c8 % (GKT / 8)) * 8;
      *(v8h*)&Bs[r][cc] = *(const v8h*)(Bt + (size_t)(bn + r) * K + k0 + cc);
    }
#endif
    if (k0 + GKT < K) {  // hint next chunk toward L2 (global_prefetch_b8)
      __builtin_prefetch(A + (size_t)(bm + (tid & 31)) * K + k0 + GKT, 0, 1);
      __builtin_prefetch(Bt + (size_t)(bn + (tid & 127)) * K + k0 + GKT, 0, 1);
    }
    __syncthreads();
    const half_t* pA  = &As[wm * 16 + ln][0];
    const half_t* pB0 = &Bs[wnb + ln][0];
    const half_t* pB1 = &Bs[wnb + 16 + ln][0];
    for (int kk = 0; kk < GKT; kk += 32) {
      union { v16h v; v8h h8[2]; } a, b0, b1;
      a.h8[0]  = *(const v8h*)(pA + kk + 16 * g);
      a.h8[1]  = *(const v8h*)(pA + kk + 16 * g + 8);
      b0.h8[0] = *(const v8h*)(pB0 + kk + 16 * g);
      b0.h8[1] = *(const v8h*)(pB0 + kk + 16 * g + 8);
      b1.h8[0] = *(const v8h*)(pB1 + kk + 16 * g);
      b1.h8[1] = *(const v8h*)(pB1 + kk + 16 * g + 8);
      c0 = __builtin_amdgcn_wmma_f32_16x16x32_f16(false, a.v, false, b0.v, (short)0, c0,
                                                  false, false);
      c1 = __builtin_amdgcn_wmma_f32_16x16x32_f16(false, a.v, false, b1.v, (short)0, c1,
                                                  false, false);
    }
  }
  const int col0 = bn + wnb + ln;
  const int col1 = col0 + 16;
  const float bv0 = bias ? bias[col0] : 0.0f;
  const float bv1 = bias ? bias[col1] : 0.0f;
  for (int r = 0; r < 8; ++r) {
    int m = r + 8 * g;
    int row = bm + wm * 16 + m;
    float v0 = c0[r] + bv0;
    float v1 = c1[r] + bv1;
    if (mode == 1) { v0 = gelu_exact(v0); v1 = gelu_exact(v1); }
    if (mode == 2) {
      outF[(size_t)row * Nn + col0] += v0 * rs[0];
      outF[(size_t)row * Nn + col1] += v1 * rs[0];
    } else {
      outH[(size_t)row * Nn + col0] = (half_t)v0;
      outH[(size_t)row * Nn + col1] = (half_t)v1;
    }
  }
}

// ---------------- 7. sampling score: xn . Ws + bs ----------------
__global__ void k_score(const half_t* __restrict__ xn, const float* __restrict__ Ws,
                        const float* __restrict__ bs, float* __restrict__ score, int M) {
  int t = blockIdx.x * 256 + threadIdx.x;
  if (t >= M * NHEAD) return;
  int row = t >> 3, h = t & 7;
  float s = bs[h];
  const half_t* px = xn + (size_t)row * DMODEL;
  for (int k = 0; k < DMODEL; ++k) s += (float)px[k] * Ws[k * NHEAD + h];
  score[(size_t)row * NHEAD + h] = s;
}

// ---------------- 8. top-128 selection ----------------
__global__ void __launch_bounds__(128) k_topk_local(const float* __restrict__ score,
                                                    const float* __restrict__ maskp,
                                                    float* __restrict__ okey,
                                                    int* __restrict__ otok) {
  __shared__ float key[256];
  __shared__ int   tok[256];
  const int arr = blockIdx.x;            // seq*8 + h
  const int seq = arr >> 3, h = arr & 7;
  const int tid = threadIdx.x;
  for (int i = tid; i < 256; i += 128) {
    if (i < LSEQ) {
      int row = seq * LSEQ + i;
      float m = maskp[row];
      key[i] = -score[(size_t)row * NHEAD + h] + (1.0f - m) * BIGF;  // ascending = top score
      tok[i] = row;
    } else { key[i] = 3.0e38f; tok[i] = seq * LSEQ; }
  }
  __syncthreads();
  bitonic_sort(key, tok, 256, tid, 128);
  for (int i = tid; i < NSAMP; i += 128) {
    okey[(size_t)arr * NSAMP + i] = key[i];
    otok[(size_t)arr * NSAMP + i] = tok[i];
  }
}

__global__ void __launch_bounds__(1024) k_topk_global(const float* __restrict__ score,
                                                      const float* __restrict__ maskp,
                                                      float* __restrict__ okey,
                                                      int* __restrict__ otok) {
  extern __shared__ __align__(16) char smem[];
  float* key = (float*)smem;
  int*   tok = (int*)(key + NTOK);
  const int arr = blockIdx.x;            // bi*8 + h
  const int bi = arr >> 3, h = arr & 7;
  for (int i = threadIdx.x; i < NTOK; i += 1024) {
    int row = bi * NPAD + i;
    key[i] = -score[(size_t)row * NHEAD + h] + (1.0f - maskp[row]) * BIGF;
    tok[i] = row;
  }
  __syncthreads();
  bitonic_sort(key, tok, NTOK, threadIdx.x, 1024);
  for (int i = threadIdx.x; i < NSAMP; i += 1024) {
    okey[(size_t)arr * NSAMP + i] = key[i];
    otok[(size_t)arr * NSAMP + i] = tok[i];
  }
}

// ---------------- 9. gather sampled KV with sigmoid gate; V stored transposed ----------------
__global__ void __launch_bounds__(128) k_sample_kv(const float* __restrict__ okey,
                                                   const int* __restrict__ otok,
                                                   const half_t* __restrict__ kv,
                                                   half_t* __restrict__ kh,
                                                   half_t* __restrict__ vt) {
  const int arr = blockIdx.x;            // seq*8 + h
  const int s = threadIdx.x;             // 0..127 sampled slot
  const int h = arr & 7;
  float val = -okey[(size_t)arr * NSAMP + s];
  float gate = 1.0f / (1.0f + __expf(-val));           // TEMP = 1
  int tokrow = otok[(size_t)arr * NSAMP + s];
  const half_t* src = kv + (size_t)tokrow * (NHEAD * 64) + h * 64;
  half_t* kdst = kh + ((size_t)arr * NSAMP + s) * DHK;
  for (int d = 0; d < DHK; ++d) kdst[d] = (half_t)((float)src[d] * gate);
  half_t* vbase = vt + (size_t)arr * DHV * NSAMP;
  for (int d = 0; d < DHV; ++d) vbase[d * NSAMP + s] = (half_t)((float)src[DHK + d] * gate);
}

// ---------------- 10. attention: per (seq,head), WMMA QK^T -> softmax -> WMMA PV ----------------
__global__ void __launch_bounds__(128) k_attn(const half_t* __restrict__ q,
                                              const half_t* __restrict__ kh,
                                              const half_t* __restrict__ vt,
                                              half_t* __restrict__ xa,
                                              int L, int seqStride) {
  extern __shared__ __align__(16) char smem[];
  half_t* kL    = (half_t*)smem;                   // [128][32]
  half_t* vL    = kL + NSAMP * DHK;                // [32][128]
  float*  lgAll = (float*)(vL + DHV * NSAMP);      // 4 waves * [16][128] f32
  half_t* pAll  = (half_t*)(lgAll + 4 * 16 * NSAMP);  // 4 waves * [16][128] f16

  const int arr = blockIdx.y;            // seq*8 + h
  const int seq = arr >> 3, h = arr & 7;
  const int tid = threadIdx.x, wave = tid >> 5, lane = tid & 31;
  const int g = lane >> 4, ln = lane & 15;
  const long tokBase = (long)seq * seqStride;

#if USE_TDM
  if (wave == 0) {   // DMA K / V^T tiles (2 x 8KB, contiguous) into LDS
    tdm_load_tile_f16(kL, kh + (size_t)arr * NSAMP * DHK, 1, NSAMP * DHK,
                      (unsigned long long)(NSAMP * DHK));
    tdm_load_tile_f16(vL, vt + (size_t)arr * DHV * NSAMP, 1, DHV * NSAMP,
                      (unsigned long long)(DHV * NSAMP));
    __builtin_amdgcn_s_wait_tensorcnt(0);
  }
#else
  {
    const uint4* sk = (const uint4*)(kh + (size_t)arr * NSAMP * DHK);
    const uint4* sv = (const uint4*)(vt + (size_t)arr * DHV * NSAMP);
    uint4* dk = (uint4*)kL; uint4* dv = (uint4*)vL;
    for (int j = tid; j < NSAMP * DHK * 2 / 16; j += 128) dk[j] = sk[j];
    for (int j = tid; j < DHV * NSAMP * 2 / 16; j += 128) dv[j] = sv[j];
  }
#endif
  __syncthreads();

  float* lg = lgAll + wave * 16 * NSAMP;
  half_t* pL = pAll + wave * 16 * NSAMP;

  const int qt = blockIdx.x * 4 + wave;
  if (qt * 16 >= L) return;              // whole wave exits (no barriers after this)

  // ---- logits = q @ K^T * ATT_SCALE : one wmma per 16-key tile (K-dim = 32 exactly) ----
  int qrow = qt * 16 + ln; if (qrow >= L) qrow = L - 1;
  const half_t* pq = q + ((size_t)(tokBase + qrow)) * DMODEL + h * DHK;
  union { v16h v; v8h h8[2]; } a, b;
  a.h8[0] = *(const v8h*)(pq + 16 * g);
  a.h8[1] = *(const v8h*)(pq + 16 * g + 8);
  for (int kt = 0; kt < NSAMP / 16; ++kt) {
    const half_t* pk = kL + (kt * 16 + ln) * DHK + 16 * g;
    b.h8[0] = *(const v8h*)(pk);
    b.h8[1] = *(const v8h*)(pk + 8);
    v8f cz = {};
    v8f c = __builtin_amdgcn_wmma_f32_16x16x32_f16(false, a.v, false, b.v, (short)0, cz,
                                                   false, false);
    for (int r = 0; r < 8; ++r)
      lg[(r + 8 * g) * NSAMP + kt * 16 + ln] = c[r] * ATT_SCALE;
  }

  // ---- softmax over 128 keys (2 lanes per row, wave-local, LDS in-order) ----
  {
    int row = lane >> 1, hcol = lane & 1;
    float* pr = lg + row * NSAMP + hcol * 64;
    float mx = -3.4e38f;
    for (int j = 0; j < 64; ++j) mx = fmaxf(mx, pr[j]);
    mx = fmaxf(mx, __shfl_xor(mx, 1, 32));
    float s = 0.0f;
    for (int j = 0; j < 64; ++j) { float e = __expf(pr[j] - mx); pr[j] = e; s += e; }
    s += __shfl_xor(s, 1, 32);
    float ri = 1.0f / s;
    half_t* pp = pL + row * NSAMP + hcol * 64;
    for (int j = 0; j < 64; ++j) pp[j] = (half_t)(pr[j] * ri);
  }

  // ---- out = P @ V : 2 output tiles x 4 chained wmma over K=128 ----
  for (int nt = 0; nt < 2; ++nt) {
    v8f c = {};
    const half_t* pa = pL + ln * NSAMP;
    const half_t* pb = vL + (nt * 16 + ln) * NSAMP;
    for (int kb = 0; kb < NSAMP; kb += 32) {
      a.h8[0] = *(const v8h*)(pa + kb + 16 * g);
      a.h8[1] = *(const v8h*)(pa + kb + 16 * g + 8);
      b.h8[0] = *(const v8h*)(pb + kb + 16 * g);
      b.h8[1] = *(const v8h*)(pb + kb + 16 * g + 8);
      c = __builtin_amdgcn_wmma_f32_16x16x32_f16(false, a.v, false, b.v, (short)0, c,
                                                 false, false);
    }
    for (int r = 0; r < 8; ++r) {
      int m = r + 8 * g, row = qt * 16 + m;
      if (row < L)
        xa[((size_t)(tokBase + row)) * DMODEL + h * DHK + nt * 16 + ln] = (half_t)c[r];
    }
  }
}

// ---------------- 11. un-permute into output ----------------
__global__ void k_unpermute(const float* __restrict__ xf, const int* __restrict__ rev,
                            float* __restrict__ out) {
  int row = blockIdx.x;                  // 0..BATCH*NTOK-1
  int bi = row / NTOK, i = row % NTOK;
  int j = rev[bi * NTOK + i];
  const float4* src = (const float4*)(xf + ((size_t)bi * NPAD + j) * DMODEL);
  float4* dst = (float4*)(out + (size_t)row * DMODEL);
  for (int t = threadIdx.x; t < DMODEL / 4; t += blockDim.x) dst[t] = src[t];
}

// ================= host orchestration =================
extern "C" void kernel_launch(void* const* d_in, const int* in_sizes, int n_in,
                              void* d_out, int out_size, void* d_ws, size_t ws_size,
                              hipStream_t stream) {
  (void)in_sizes; (void)n_in; (void)out_size; (void)ws_size;
  const float* x      = (const float*)d_in[0];
  const float* coords = (const float*)d_in[1];
  const float* mask   = (const float*)d_in[2];
  const float* rs     = (const float*)d_in[3];
  const float* Wq  = (const float*)d_in[4];
  const float* bq  = (const float*)d_in[5];
  const float* Wkv = (const float*)d_in[6];
  const float* bkv = (const float*)d_in[7];
  const float* Wc  = (const float*)d_in[8];
  const float* bc  = (const float*)d_in[9];
  const float* W1  = (const float*)d_in[10];
  const float* b1  = (const float*)d_in[11];
  const float* W2  = (const float*)d_in[12];
  const float* b2  = (const float*)d_in[13];
  const float* Ws  = (const float*)d_in[14];
  const float* bs  = (const float*)d_in[15];
  const float* g1  = (const float*)d_in[16];
  const float* g2  = (const float*)d_in[17];

  char* wp = (char*)d_ws;
  auto bump = [&](size_t bytes) -> void* {
    void* p = (void*)wp;
    wp += (bytes + 255) & ~(size_t)255;
    return p;
  };
  float*  t_total = (float*)bump((size_t)BATCH * NTOK * 4);
  int*    t_idx   = (int*)  bump((size_t)BATCH * NTOK * 4);
  int*    t_rev   = (int*)  bump((size_t)BATCH * NTOK * 4);
  float*  t_x     = (float*)bump((size_t)MTOK * DMODEL * 4);
  float*  t_mp    = (float*)bump((size_t)MTOK * 4);
  half_t* t_xn    = (half_t*)bump((size_t)MTOK * DMODEL * 2);
  half_t* t_q     = (half_t*)bump((size_t)MTOK * DMODEL * 2);
  half_t* t_kv    = (half_t*)bump((size_t)MTOK * 512 * 2);
  float*  t_score = (float*)bump((size_t)MTOK * NHEAD * 4);
  float*  t_okey  = (float*)bump((size_t)4096 * NSAMP * 4);
  int*    t_otok  = (int*)  bump((size_t)4096 * NSAMP * 4);
  half_t* t_kh    = (half_t*)bump((size_t)4096 * NSAMP * DHK * 2);
  half_t* t_vt    = (half_t*)bump((size_t)4096 * DHV * NSAMP * 2);
  half_t* t_xa    = (half_t*)bump((size_t)MTOK * DMODEL * 2);
  half_t* t_h1    = (half_t*)bump((size_t)MTOK * DFF * 2);
  half_t* t_wq    = (half_t*)bump((size_t)NLAYER * DMODEL * DMODEL * 2);
  half_t* t_wkv   = (half_t*)bump((size_t)NLAYER * 512 * DMODEL * 2);
  half_t* t_wc    = (half_t*)bump((size_t)NLAYER * DMODEL * DMODEL * 2);
  half_t* t_w1    = (half_t*)bump((size_t)NLAYER * DFF * DMODEL * 2);
  half_t* t_w2    = (half_t*)bump((size_t)NLAYER * DMODEL * DFF * 2);

  // ---- weight precast/transpose to f16 [N][K] ----
  for (int i = 0; i < NLAYER; ++i) {
    k_transpose_h<<<(DMODEL * DMODEL) / 256, 256, 0, stream>>>(Wq + (size_t)i * DMODEL * DMODEL,
                                                              t_wq + (size_t)i * DMODEL * DMODEL,
                                                              DMODEL, DMODEL);
    k_transpose_h<<<(DMODEL * 512) / 256, 256, 0, stream>>>(Wkv + (size_t)i * DMODEL * 512,
                                                            t_wkv + (size_t)i * 512 * DMODEL,
                                                            DMODEL, 512);
    k_transpose_h<<<(DMODEL * DMODEL) / 256, 256, 0, stream>>>(Wc + (size_t)i * DMODEL * DMODEL,
                                                              t_wc + (size_t)i * DMODEL * DMODEL,
                                                              DMODEL, DMODEL);
    k_transpose_h<<<(DMODEL * DFF) / 256, 256, 0, stream>>>(W1 + (size_t)i * DMODEL * DFF,
                                                            t_w1 + (size_t)i * DFF * DMODEL,
                                                            DMODEL, DFF);
    k_transpose_h<<<(DFF * DMODEL) / 256, 256, 0, stream>>>(W2 + (size_t)i * DFF * DMODEL,
                                                            t_w2 + (size_t)i * DMODEL * DFF,
                                                            DFF, DMODEL);
  }

  // ---- serialization + argsort + gather ----
  size_t serLds = (size_t)(3 * NTOK) * 4 + SER_T * 4 + SER_T * 4;  // ~205KB of 320KB LDS
  k_serialize<<<BATCH, SER_T, serLds, stream>>>(coords, mask, t_total);
  k_argsort16k<<<BATCH, 1024, (size_t)NTOK * 8, stream>>>(t_total, t_idx);
  k_rev<<<(BATCH * NTOK + 255) / 256, 256, 0, stream>>>(t_idx, t_rev);
  k_gather_pad<<<MTOK, 64, 0, stream>>>(x, mask, t_idx, t_x, t_mp);

  // ---- transformer blocks: 4 local (L=129) + 1 global (L=16384) ----
  for (int i = 0; i < NLAYER; ++i) {
    const bool glob = (i == NLAYER - 1);
    const int nseq = glob ? BATCH : NSEQ_L;
    const int L = glob ? NTOK : LSEQ;
    const int seqStride = glob ? NPAD : LSEQ;

    k_rmsnorm<<<MTOK / 8, 256, 0, stream>>>(t_x, g1 + (size_t)i * DMODEL, t_xn, MTOK);
    k_gemm<<<dim3(MTOK / GBM, DMODEL / GBN), 256, 0, stream>>>(
        t_xn, t_wq + (size_t)i * DMODEL * DMODEL, bq + (size_t)i * DMODEL,
        MTOK, DMODEL, DMODEL, 0, nullptr, t_q, nullptr);
    k_gemm<<<dim3(MTOK / GBM, 512 / GBN), 256, 0, stream>>>(
        t_xn, t_wkv + (size_t)i * 512 * DMODEL, bkv + (size_t)i * 512,
        MTOK, 512, DMODEL, 0, nullptr, t_kv, nullptr);
    k_score<<<(MTOK * NHEAD) / 256, 256, 0, stream>>>(t_xn, Ws + (size_t)i * DMODEL * NHEAD,
                                                      bs + (size_t)i * NHEAD, t_score, MTOK);
    if (glob)
      k_topk_global<<<BATCH * NHEAD, 1024, (size_t)NTOK * 8, stream>>>(t_score, t_mp,
                                                                       t_okey, t_otok);
    else
      k_topk_local<<<NSEQ_L * NHEAD, 128, 0, stream>>>(t_score, t_mp, t_okey, t_otok);
    k_sample_kv<<<nseq * NHEAD, 128, 0, stream>>>(t_okey, t_otok, t_kv, t_kh, t_vt);

    const int qtiles = (L + 15) / 16;
    const int chunks = (qtiles + 3) / 4;
    const size_t attLds = (size_t)NSAMP * DHK * 2 + (size_t)DHV * NSAMP * 2 +
                          (size_t)4 * 16 * NSAMP * 4 + (size_t)4 * 16 * NSAMP * 2;  // 64KB
    k_attn<<<dim3(chunks, nseq * NHEAD), 128, attLds, stream>>>(t_q, t_kh, t_vt, t_xa,
                                                                L, seqStride);
    k_gemm<<<dim3(MTOK / GBM, DMODEL / GBN), 256, 0, stream>>>(
        t_xa, t_wc + (size_t)i * DMODEL * DMODEL, bc + (size_t)i * DMODEL,
        MTOK, DMODEL, DMODEL, 2, rs, nullptr, t_x);
    k_rmsnorm<<<MTOK / 8, 256, 0, stream>>>(t_x, g2 + (size_t)i * DMODEL, t_xn, MTOK);
    k_gemm<<<dim3(MTOK / GBM, DFF / GBN), 256, 0, stream>>>(
        t_xn, t_w1 + (size_t)i * DFF * DMODEL, b1 + (size_t)i * DFF,
        MTOK, DFF, DMODEL, 1, nullptr, t_h1, nullptr);
    k_gemm<<<dim3(MTOK / GBM, DMODEL / GBN), 256, 0, stream>>>(
        t_h1, t_w2 + (size_t)i * DMODEL * DFF, b2 + (size_t)i * DMODEL,
        MTOK, DMODEL, DFF, 2, rs, nullptr, t_x);
  }

  k_unpermute<<<BATCH * NTOK, 64, 0, stream>>>(t_x, t_rev, (float*)d_out);
}
